// VoxelBackbone8x_29317446762674
// MI455X (gfx1250) — matrix operand order repaired and checked by reading, not compiled, not run on hardware
//
#include <hip/hip_runtime.h>

// ---------- feature probes ----------
#if defined(__has_builtin)
#if __has_builtin(__builtin_amdgcn_global_load_async_to_lds_b128)
#define HAVE_ASYNC_LDS 1
#endif
#if __has_builtin(__builtin_amdgcn_s_wait_asynccnt)
#define HAVE_WAIT_ASYNC 1
#endif
#endif

// ---------- types ----------
typedef __attribute__((ext_vector_type(16))) __bf16 v16bf;
typedef __attribute__((ext_vector_type(8)))  float  v8f;
typedef __attribute__((ext_vector_type(4)))  int    v4i;
typedef unsigned short bf16_t;   // bf16 storage bits (host-safe)

#define AS_GLOBAL __attribute__((address_space(1)))
#define AS_LOCAL  __attribute__((address_space(3)))

struct __align__(16) U4 { unsigned x, y, z, w; };
struct __align__(8)  U2 { unsigned x, y; };

// ---------- device helpers ----------
__device__ __forceinline__ bf16_t f2bf(float f) {
  unsigned int u = __float_as_uint(f);
  unsigned int r = (u + 0x7FFFu + ((u >> 16) & 1u)) >> 16;
  return (bf16_t)r;
}

// ---------- small kernels ----------
__global__ void mask_init_k(const float* __restrict__ logits, float* __restrict__ mask, int n) {
  int i = blockIdx.x * blockDim.x + threadIdx.x;
  if (i < n) mask[i] = (logits[i] < 0.05f) ? 1.0f : 0.0f;
}

__global__ void x_init_k(const float* __restrict__ x, const float* __restrict__ mask,
                         bf16_t* __restrict__ xout, int ntot) {
  int i = blockIdx.x * blockDim.x + threadIdx.x;
  if (i < ntot) xout[i] = f2bf(x[i] * mask[i >> 2]);   // C_in = 4
}

__global__ void zero_k(float* __restrict__ p, int n) {
  int i = blockIdx.x * blockDim.x + threadIdx.x;
  if (i < n) p[i] = 0.0f;
}

__global__ void mask_count_k(const float* __restrict__ mask, float* __restrict__ cnt, int n) {
  __shared__ float red[256];
  int i = blockIdx.x * 256 + threadIdx.x;
  red[threadIdx.x] = (i < n) ? mask[i] : 0.0f;
  __syncthreads();
  for (int s = 128; s > 0; s >>= 1) {
    if ((int)threadIdx.x < s) red[threadIdx.x] += red[threadIdx.x + s];
    __syncthreads();
  }
  if (threadIdx.x == 0) atomicAdd(cnt, red[0]);
}

__global__ void mask_pool_k(const float* __restrict__ mi, float* __restrict__ mo,
                            int Di, int Hi, int Wi, int Do, int Ho, int Wo,
                            int sd, int sh, int sw, int n) {
  int i = blockIdx.x * blockDim.x + threadIdx.x;
  if (i >= n) return;
  int xo = i % Wo; int t = i / Wo;
  int yo = t % Ho; t /= Ho;
  int zo = t % Do; int b = t / Do;
  float v = 0.0f;
  for (int dz = 0; dz < 3; ++dz) {
    int zi = zo * sd + dz - 1;
    if ((unsigned)zi >= (unsigned)Di) continue;
    for (int dy = 0; dy < 3; ++dy) {
      int yi = yo * sh + dy - 1;
      if ((unsigned)yi >= (unsigned)Hi) continue;
      for (int dx = 0; dx < 3; ++dx) {
        int xi = xo * sw + dx - 1;
        if ((unsigned)xi >= (unsigned)Wi) continue;
        v = fmaxf(v, mi[((size_t)(b * Di + zi) * Hi + yi) * Wi + xi]);
      }
    }
  }
  mo[i] = (v > 0.0f) ? 1.0f : 0.0f;
}

// Pre-swizzle fp32 weights [K][Co] into wave32 B-fragment order:
// frag idx ((kc*NT+nt)*32+lane)*16 + j ; n = nt*16+(lane&15), K = kc*32 + (j>>3)*16 + (lane>>4)*8 + (j&7)
__global__ void wprep_k(const float* __restrict__ W, bf16_t* __restrict__ wf,
                        int K, int Co, int KC, int NT) {
  int t = blockIdx.x * blockDim.x + threadIdx.x;
  int total = KC * NT * 32;
  if (t >= total) return;
  int lane = t & 31;
  int nt   = (t >> 5) % NT;
  int kc   = t / (32 * NT);
  int n    = nt * 16 + (lane & 15);
  int half = lane >> 4;
  bf16_t* dst = wf + (size_t)t * 16;
#pragma unroll
  for (int j = 0; j < 16; ++j) {
    int k = kc * 32 + (j >> 3) * 16 + half * 8 + (j & 7);
    float w = (k < K) ? W[(size_t)k * Co + n] : 0.0f;
    dst[j] = f2bf(w);
  }
}

// ---------- implicit-GEMM conv via WMMA bf16 ----------
// Block = 8 waves; one wave: 16 output sites x (NT*16) channels; K = 27*Ci padded to 32.
// B fragments for the current K-chunk are staged into LDS (async if available),
// double-buffered, and shared by all 8 waves.
template<int NT>
__global__ __launch_bounds__(256) void conv_wmma_k(
    const bf16_t* __restrict__ xin, const bf16_t* __restrict__ wf,
    float* __restrict__ y, const float* __restrict__ mask,
    float* __restrict__ stats,
    int Di, int Hi, int Wi, int ciShift,
    int Do, int Ho, int Wo, int Co,
    int sd, int sh, int sw, int K, int KC, int Msites) {
  __shared__ __align__(16) bf16_t bsh[2][NT * 512];

  const int tid   = threadIdx.x;
  const int wave  = tid >> 5;
  const int lane  = tid & 31;
  const int mtile = blockIdx.x * 8 + wave;
  const int mbase = mtile * 16;
  const bool waveActive = (mbase < Msites);

  const int Ci   = 1 << ciShift;
  const int half = lane >> 4;

  const int mA = mbase + (lane & 15);
  const bool mAvalid = (mA < Msites);
  int sA = mAvalid ? mA : 0;
  int xo = sA % Wo; int t = sA / Wo;
  int yo = t % Ho;  t /= Ho;
  int zo = t % Do;  int b = t / Do;
  const int zb = zo * sd - 1, yb = yo * sh - 1, xb = xo * sw - 1;
  const int HiWi = Hi * Wi;
  const size_t baseB = (size_t)b * Di * HiWi;

  v8f acc[NT];
#pragma unroll
  for (int nt = 0; nt < NT; ++nt) {
    v8f z = {0.f,0.f,0.f,0.f,0.f,0.f,0.f,0.f};
    acc[nt] = z;
  }

  auto stageChunk = [&](int kc, int buf) {
    if (tid < NT * 64) {                       // NT*64 x 16B = NT KB chunk
      const U4* gsrc = (const U4*)(wf + (size_t)kc * NT * 512) + tid;
      U4* ldst = (U4*)(&bsh[buf][0]) + tid;
#if HAVE_ASYNC_LDS
      U4* gnc = const_cast<U4*>(gsrc);
      __builtin_amdgcn_global_load_async_to_lds_b128(
          (AS_GLOBAL v4i*)gnc, (AS_LOCAL v4i*)ldst, 0, 0);
#else
      *ldst = *gsrc;
#endif
    }
  };
  stageChunk(0, 0);

  for (int kc = 0; kc < KC; ++kc) {
    const int buf = kc & 1;
#if HAVE_WAIT_ASYNC
    __builtin_amdgcn_s_wait_asynccnt(0);
#endif
    __syncthreads();
    if (kc + 1 < KC) stageChunk(kc + 1, buf ^ 1);

    // ---- A fragment: two (or four) vectorized runs of K-consecutive elements.
    // elems j<8: K = kc*32 + 8*half + j ; elems j>=8: K = kc*32 + 16 + 8*half + (j-8)
    union AU { bf16_t u[16]; U2 d2[4]; U4 q[2]; v16bf v; } au;
    if (ciShift >= 3) {          // Ci >= 8: each 8-run = one tap, contiguous channels
#pragma unroll
      for (int g = 0; g < 2; ++g) {
        int k0 = kc * 32 + g * 16 + half * 8;
        U4 q = {0u, 0u, 0u, 0u};
        if (mAvalid && k0 < K) {
          int tap = k0 >> ciShift;
          int c   = k0 & (Ci - 1);
          int dz = tap / 9; int r9 = tap - dz * 9;
          int dy = r9 / 3;  int dx = r9 - dy * 3;
          int zi = zb + dz, yi = yb + dy, xi = xb + dx;
          if ((unsigned)zi < (unsigned)Di && (unsigned)yi < (unsigned)Hi &&
              (unsigned)xi < (unsigned)Wi)
            q = *(const U4*)(xin + (baseB + (size_t)zi * HiWi + (size_t)yi * Wi + xi) * Ci + c);
        }
        au.q[g] = q;
      }
    } else {                     // Ci == 4: 4-element runs, 8B loads
#pragma unroll
      for (int g = 0; g < 4; ++g) {
        int k0 = kc * 32 + (g >> 1) * 16 + half * 8 + (g & 1) * 4;
        U2 d = {0u, 0u};
        if (mAvalid && k0 < K) {
          int tap = k0 >> 2;
          int dz = tap / 9; int r9 = tap - dz * 9;
          int dy = r9 / 3;  int dx = r9 - dy * 3;
          int zi = zb + dz, yi = yb + dy, xi = xb + dx;
          if ((unsigned)zi < (unsigned)Di && (unsigned)yi < (unsigned)Hi &&
              (unsigned)xi < (unsigned)Wi)
            d = *(const U2*)(xin + (baseB + (size_t)zi * HiWi + (size_t)yi * Wi + xi) * 4);
        }
        au.d2[g] = d;
      }
    }
    v16bf a = au.v;

#pragma unroll
    for (int nt = 0; nt < NT; ++nt) {
      v16bf bfr = *(const v16bf*)&bsh[buf][(nt * 32 + lane) * 16];
      acc[nt] = __builtin_amdgcn_wmma_f32_16x16x32_bf16(
          false, a, false, bfr, (short)0, acc[nt], false, false);
    }
  }

  // ---- epilogue: store y (fp32) + masked BN partial sums
  if (waveActive) {
#pragma unroll
    for (int nt = 0; nt < NT; ++nt) {
      const int n = nt * 16 + (lane & 15);
      float s = 0.0f, sq = 0.0f;
#pragma unroll
      for (int r = 0; r < 8; ++r) {
        int srow = mbase + r + 8 * half;   // C/D layout: VGPR r -> M = r + 8*(lane>=16)
        if (srow < Msites) {
          float v = acc[nt][r];
          y[(size_t)srow * Co + n] = v;
          float mk = mask[srow];
          s  += v * mk;
          sq += v * v * mk;
        }
      }
      s  += __shfl_xor(s,  16, 32);
      sq += __shfl_xor(sq, 16, 32);
      if (half == 0) {
        atomicAdd(&stats[n],       s);
        atomicAdd(&stats[128 + n], sq);
      }
    }
  }
}

// ---------- BN(active-voxel stats) + ReLU + mask ----------
__global__ void bn_apply_k(const float* __restrict__ y, const float* __restrict__ mask,
                           const float* __restrict__ stats,
                           const float* __restrict__ gamma, const float* __restrict__ beta,
                           bf16_t* __restrict__ xout, float* __restrict__ fout, int writeFloat,
                           int Msites, int Co) {
  size_t i = (size_t)blockIdx.x * blockDim.x + threadIdx.x;
  size_t tot = (size_t)Msites * Co;
  if (i >= tot) return;
  int n = (int)(i % Co);
  size_t s = i / Co;
  float cnt  = fmaxf(stats[256], 1.0f);
  float mean = stats[n] / cnt;
  float var  = stats[128 + n] / cnt - mean * mean;
  float inv  = rsqrtf(var + 1e-5f);
  float v = (y[i] - mean) * inv * gamma[n] + beta[n];
  v = fmaxf(v, 0.0f) * mask[s];
  if (writeFloat) fout[i] = v;
  else            xout[i] = f2bf(v);
}

// ---------- host orchestration ----------
extern "C" void kernel_launch(void* const* d_in, const int* in_sizes, int n_in,
                              void* d_out, int out_size, void* d_ws, size_t ws_size,
                              hipStream_t stream) {
  (void)in_sizes; (void)n_in; (void)out_size; (void)ws_size;

  const float* x      = (const float*)d_in[0];
  const float* logits = (const float*)d_in[1];

  char* ws = (char*)d_ws;
  bf16_t* xbufA = (bf16_t*)(ws + 0x0000000ULL);   // 42 MB
  bf16_t* xbufB = (bf16_t*)(ws + 0x2800000ULL);   // 42 MB
  float*  ybuf  = (float*) (ws + 0x5000000ULL);   // 84 MB
  float*  maskA = (float*) (ws + 0xA000000ULL);   // 6 MB
  float*  maskB = (float*) (ws + 0xA600000ULL);   // 6 MB
  float*  stats = (float*) (ws + 0xAC00000ULL);   // 257 floats
  bf16_t* wfrag = (bf16_t*)(ws + 0xAC10000ULL);   // ~1 MB swizzled weights (reused per layer)

  int Di = 40, Hi = 128, Wi = 128;
  const int Bn = 2;
  int nsites0 = Bn * Di * Hi * Wi;

  mask_init_k<<<dim3((nsites0 + 255) / 256), dim3(256), 0, stream>>>(logits, maskA, nsites0);
  x_init_k<<<dim3((nsites0 * 4 + 255) / 256), dim3(256), 0, stream>>>(x, maskA, xbufA, nsites0 * 4);

  // {sp?, Ci, Co, sd, sh, sw, log2(Ci)}
  static const int cfg[12][7] = {
    {0,  4, 16, 1,1,1, 2}, {1, 16, 32, 2,2,2, 4}, {0, 32, 32, 1,1,1, 5}, {0, 32, 32, 1,1,1, 5},
    {1, 32, 64, 2,2,2, 5}, {0, 64, 64, 1,1,1, 6}, {0, 64, 64, 1,1,1, 6},
    {1, 64,128, 2,2,2, 6}, {0,128,128, 1,1,1, 7}, {0,128,128, 1,1,1, 7},
    {1,128,128, 1,2,2, 7}, {0,128,128, 1,1,1, 7}};

  bf16_t* xin  = xbufA; bf16_t* xout = xbufB;
  float*  mcur = maskA; float*  mnxt = maskB;

  for (int l = 0; l < 12; ++l) {
    const int sp = cfg[l][0], Ci = cfg[l][1], Co = cfg[l][2];
    const int sd = cfg[l][3], sh = cfg[l][4], sw = cfg[l][5], cish = cfg[l][6];
    const float* Wp = (const float*)d_in[2 + 3 * l];
    const float* gm = (const float*)d_in[3 + 3 * l];
    const float* bt = (const float*)d_in[4 + 3 * l];

    int Do = Di, Ho = Hi, Wo = Wi;
    float* mout = mcur;
    if (sp) {
      Do = (Di - 1) / sd + 1;
      Ho = (Hi - 1) / sh + 1;
      Wo = (Wi - 1) / sw + 1;
      mout = mnxt;
    }
    const int Ms = Bn * Do * Ho * Wo;
    if (sp)
      mask_pool_k<<<dim3((Ms + 255) / 256), dim3(256), 0, stream>>>(
          mcur, mout, Di, Hi, Wi, Do, Ho, Wo, sd, sh, sw, Ms);

    zero_k<<<dim3(2), dim3(256), 0, stream>>>(stats, 257);
    mask_count_k<<<dim3((Ms + 255) / 256), dim3(256), 0, stream>>>(mout, stats + 256, Ms);

    const int K = 27 * Ci, KC = (K + 31) / 32, NT = Co / 16;
    wprep_k<<<dim3((KC * NT * 32 + 255) / 256), dim3(256), 0, stream>>>(Wp, wfrag, K, Co, KC, NT);

    const int Mtiles = (Ms + 15) / 16;
    dim3 grid((Mtiles + 7) / 8), blk(256);
    switch (NT) {
      case 1: conv_wmma_k<1><<<grid, blk, 0, stream>>>(xin, wfrag, ybuf, mout, stats,
                  Di, Hi, Wi, cish, Do, Ho, Wo, Co, sd, sh, sw, K, KC, Ms); break;
      case 2: conv_wmma_k<2><<<grid, blk, 0, stream>>>(xin, wfrag, ybuf, mout, stats,
                  Di, Hi, Wi, cish, Do, Ho, Wo, Co, sd, sh, sw, K, KC, Ms); break;
      case 4: conv_wmma_k<4><<<grid, blk, 0, stream>>>(xin, wfrag, ybuf, mout, stats,
                  Di, Hi, Wi, cish, Do, Ho, Wo, Co, sd, sh, sw, K, KC, Ms); break;
      default: conv_wmma_k<8><<<grid, blk, 0, stream>>>(xin, wfrag, ybuf, mout, stats,
                  Di, Hi, Wi, cish, Do, Ho, Wo, Co, sd, sh, sw, K, KC, Ms); break;
    }

    const long long tot = (long long)Ms * Co;
    bn_apply_k<<<dim3((unsigned)((tot + 255) / 256)), dim3(256), 0, stream>>>(
        ybuf, mout, stats, gm, bt, xout, (float*)d_out, (l == 11) ? 1 : 0, Ms, Co);

    bf16_t* tx = xin; xin = xout; xout = tx;
    if (sp) { float* tm = mcur; mcur = mnxt; mnxt = tm; Di = Do; Hi = Ho; Wi = Wo; }
  }
}
